// SAGEContextNodeClassifier_26731876451134
// MI455X (gfx1250) — matrix operand order repaired and checked by reading, not compiled, not run on hardware
//
#include <hip/hip_runtime.h>

typedef __attribute__((ext_vector_type(16))) __bf16 v16bf;
typedef __attribute__((ext_vector_type(4)))  __bf16 v4bf;
typedef __attribute__((ext_vector_type(8)))  float  v8f;

// ---------------------------------------------------------------------------
// WMMA helpers (V_WMMA_F32_16X16X32_BF16, wave32)
//
// A (16x32 MxK) per-lane layout: lane = ((kk>>3)&1)*16 + m ; packed element
// index e = (kk&7) | ((kk&16)>>1), kk = k within 32-wide chunk.
// Activations are staged in LDS already in this layout (bf16), so the GEMM
// A-operand fetch is a single 32-byte conflict-free LDS read.
// ---------------------------------------------------------------------------
__device__ __forceinline__ v16bf load_A(const __bf16* apack, int kc, int lane) {
    return *(const v16bf*)&apack[(kc * 32 + lane) * 16];
}

// B (32x16 KxN): pre-packed in global so each lane reads one contiguous v16bf.
__device__ __forceinline__ v16bf load_B(const v16bf* p, int ntile, int numKc, int kc, int lane) {
    return p[((ntile * numKc) + kc) * 32 + lane];
}

__device__ __forceinline__ v8f wmma_bf16(v16bf a, v16bf b, v8f c) {
    return __builtin_amdgcn_wmma_f32_16x16x32_bf16(false, a, false, b, (short)0, c, false, false);
}

// Scatter one float4 (tile row r, col base c4, c4 % 4 == 0) into the packed
// bf16 A layout. 4 consecutive cols map to 4 consecutive packed elements.
__device__ __forceinline__ void store_A4(__bf16* apack, int r, int c4, float4 v) {
    int kc    = c4 >> 5;
    int kk    = c4 & 31;
    int laneA = (((kk >> 3) & 1) << 4) | r;
    int e0    = (kk & 7) | ((kk & 16) >> 1);
    v4bf b4;
    b4[0] = (__bf16)v.x; b4[1] = (__bf16)v.y; b4[2] = (__bf16)v.z; b4[3] = (__bf16)v.w;
    *(v4bf*)&apack[((kc * 32 + laneA) << 4) + e0] = b4;
}

// Scatter a single value into the packed bf16 A layout (used for ctx mid tile).
__device__ __forceinline__ void store_A1(__bf16* apack, int r, int c, float v) {
    int kc    = c >> 5;
    int kk    = c & 31;
    int laneA = (((kk >> 3) & 1) << 4) | r;
    int e     = (kk & 7) | ((kk & 16) >> 1);
    apack[((kc * 32 + laneA) << 4) + e] = (__bf16)v;
}

// ---------------------------------------------------------------------------
// Utility kernels
// ---------------------------------------------------------------------------
__global__ void fill_f4_kernel(float4* __restrict__ p, long long n4) {
    long long t = (long long)blockIdx.x * blockDim.x + threadIdx.x;
    if (t < n4) p[t] = make_float4(0.f, 0.f, 0.f, 0.f);
}

__global__ void fill_u32_kernel(unsigned int* __restrict__ p, int n) {
    int t = blockIdx.x * blockDim.x + threadIdx.x;
    if (t < n) p[t] = 0u;
}

__global__ void degree_kernel(const long long* __restrict__ ei, unsigned int* __restrict__ deg,
                              int E) {
    int t = blockIdx.x * blockDim.x + threadIdx.x;
    if (t >= E) return;
    long long dst = ei[(long long)E + t];
    __hip_atomic_fetch_add(&deg[dst], 1u, __ATOMIC_RELAXED, __HIP_MEMORY_SCOPE_AGENT);
}

__global__ void invdeg_kernel(const unsigned int* __restrict__ deg, float* __restrict__ invd,
                              int n) {
    int t = blockIdx.x * blockDim.x + threadIdx.x;
    if (t < n) invd[t] = 1.0f / fmaxf((float)deg[t], 1.0f);
}

// Pack row-major fp32 weight W[K,Nn] into bf16 WMMA-B lane layout.
__global__ void pack_weight_kernel(const float* __restrict__ W, __bf16* __restrict__ P,
                                   int K, int Nn) {
    int t = blockIdx.x * blockDim.x + threadIdx.x;
    int numKc  = K >> 5;
    int total  = (Nn >> 4) * numKc * 32;
    if (t >= total) return;
    int lane  = t & 31;
    int kc    = (t >> 5) % numKc;
    int ntile = (t >> 5) / numKc;
    int ncol  = ntile * 16 + (lane & 15);
    int k0    = kc * 32 + ((lane >> 4) << 3);
    v16bf v;
#pragma unroll
    for (int j = 0; j < 8; ++j) {
        v[2 * j]     = (__bf16)W[(k0 + ((2 * j) & 7) + ((j >> 2) << 4))     * Nn + ncol];
        v[2 * j + 1] = (__bf16)W[(k0 + ((2 * j) & 7) + ((j >> 2) << 4) + 1) * Nn + ncol];
    }
    ((v16bf*)P)[t] = v;
}

// ---------------------------------------------------------------------------
// Edge scatter: msg[dst,:] += h[src,:]   (one wave per edge, float4 per lane)
// h and msg (51 MB each) fit in the 192 MB L2, so atomics are L2-local.
// ---------------------------------------------------------------------------
__global__ void scatter_kernel(const float* __restrict__ hin, const long long* __restrict__ ei,
                               float* __restrict__ msg, int E) {
    long long t = (long long)blockIdx.x * blockDim.x + threadIdx.x;
    long long edge = t >> 5;
    if (edge >= E) return;
    int lane = (int)(t & 31);
    long long src = ei[edge];
    long long dst = ei[(long long)E + edge];
    const float4 v = *(const float4*)&hin[src * 128 + lane * 4];
    float* d = &msg[dst * 128 + lane * 4];
    __hip_atomic_fetch_add(d + 0, v.x, __ATOMIC_RELAXED, __HIP_MEMORY_SCOPE_AGENT);
    __hip_atomic_fetch_add(d + 1, v.y, __ATOMIC_RELAXED, __HIP_MEMORY_SCOPE_AGENT);
    __hip_atomic_fetch_add(d + 2, v.z, __ATOMIC_RELAXED, __HIP_MEMORY_SCOPE_AGENT);
    __hip_atomic_fetch_add(d + 3, v.w, __ATOMIC_RELAXED, __HIP_MEMORY_SCOPE_AGENT);
}

// ---------------------------------------------------------------------------
// Fused SAGE layer: h_out = relu(LN((msg*invd)@Wl + bl + h@Wr))
// 256 threads (8 waves) per block; block handles a 16-row tile.
// ---------------------------------------------------------------------------
#define SROW 132
__global__ void sage_layer_kernel(const float* __restrict__ hin, const float* __restrict__ msg,
                                  const float* __restrict__ invd,
                                  const __bf16* __restrict__ pWl, const __bf16* __restrict__ pWr,
                                  const float* __restrict__ bl, const float* __restrict__ gma,
                                  const float* __restrict__ bta, float* __restrict__ hout,
                                  int n) {
    __shared__ __bf16 pa[4 * 32 * 16];   // agg tile, packed A layout (4 KB)
    __shared__ __bf16 ph[4 * 32 * 16];   // h   tile, packed A layout (4 KB)
    __shared__ float  lds_o[16 * SROW];  // fp32 output tile for LayerNorm

    int tid = threadIdx.x;
    int block_row = blockIdx.x * 16;

    // Cooperative staging: convert to bf16 ONCE, directly into WMMA-A layout.
    for (int i = tid; i < 16 * 32; i += 256) {
        int r  = i >> 5;
        int c4 = (i & 31) * 4;
        int row = block_row + r;
        if (row >= n) row = n - 1;
        float4 hv = *(const float4*)&hin[(long long)row * 128 + c4];
        float4 mv = *(const float4*)&msg[(long long)row * 128 + c4];
        float  s  = invd[row];
        store_A4(ph, r, c4, hv);
        store_A4(pa, r, c4, make_float4(mv.x * s, mv.y * s, mv.z * s, mv.w * s));
    }
    __syncthreads();

    int lane = tid & 31;
    int wave = tid >> 5;       // ntile 0..7 (cols wave*16 .. wave*16+15)
    const v16bf* bl_v = (const v16bf*)pWl;
    const v16bf* br_v = (const v16bf*)pWr;

    v8f acc = {};
#pragma unroll
    for (int kc = 0; kc < 4; ++kc) {
        acc = wmma_bf16(load_A(pa, kc, lane), load_B(bl_v, wave, 4, kc, lane), acc);
        acc = wmma_bf16(load_A(ph, kc, lane), load_B(br_v, wave, 4, kc, lane), acc);
    }

    int col  = wave * 16 + (lane & 15);
    float bb = bl[col];
#pragma unroll
    for (int i = 0; i < 8; ++i) {
        int r = i + ((lane >> 4) << 3);
        lds_o[r * SROW + col] = acc[i] + bb;
    }
    __syncthreads();

    // LayerNorm + ReLU: each wave handles 2 rows, wave32 shuffle reduction.
#pragma unroll
    for (int rr = 0; rr < 2; ++rr) {
        int r   = wave * 2 + rr;
        int c0  = lane * 4;
        float v0 = lds_o[r * SROW + c0 + 0], v1 = lds_o[r * SROW + c0 + 1];
        float v2 = lds_o[r * SROW + c0 + 2], v3 = lds_o[r * SROW + c0 + 3];
        float s  = v0 + v1 + v2 + v3;
        float ss = v0 * v0 + v1 * v1 + v2 * v2 + v3 * v3;
#pragma unroll
        for (int m = 16; m >= 1; m >>= 1) {
            s  += __shfl_xor(s,  m, 32);
            ss += __shfl_xor(ss, m, 32);
        }
        float mean = s * (1.0f / 128.0f);
        float var  = ss * (1.0f / 128.0f) - mean * mean;
        float rstd = rsqrtf(var + 1e-5f);
        int row = block_row + r;
        if (row < n) {
#pragma unroll
            for (int j = 0; j < 4; ++j) {
                int c = c0 + j;
                float y = (lds_o[r * SROW + c] - mean) * rstd * gma[c] + bta[c];
                hout[(long long)row * 128 + c] = fmaxf(y, 0.0f);
            }
        }
    }
}

// ---------------------------------------------------------------------------
// ctx MLP: c = relu(ctx@W1 + b1)@W2 + b2     (128 -> 64 -> 128)
// ---------------------------------------------------------------------------
__global__ void ctx_mlp_kernel(const float* __restrict__ ctx, const __bf16* __restrict__ pW1,
                               const __bf16* __restrict__ pW2, const float* __restrict__ b1,
                               const float* __restrict__ b2, float* __restrict__ cout, int n) {
    __shared__ __bf16 pin[4 * 32 * 16];  // input tile, packed A layout
    __shared__ __bf16 pmid[2 * 32 * 16]; // mid tile (16x64), packed A layout

    int tid = threadIdx.x;
    int block_row = blockIdx.x * 16;
    for (int i = tid; i < 16 * 32; i += 256) {
        int r = i >> 5, c4 = (i & 31) * 4;
        int row = block_row + r;
        if (row >= n) row = n - 1;
        float4 v = *(const float4*)&ctx[(long long)row * 128 + c4];
        store_A4(pin, r, c4, v);
    }
    __syncthreads();

    int lane = tid & 31;
    int wave = tid >> 5;
    const v16bf* w1_v = (const v16bf*)pW1;
    const v16bf* w2_v = (const v16bf*)pW2;

    if (wave < 4) {                      // stage 1: 64 output cols = 4 ntiles
        v8f acc = {};
#pragma unroll
        for (int kc = 0; kc < 4; ++kc)
            acc = wmma_bf16(load_A(pin, kc, lane), load_B(w1_v, wave, 4, kc, lane), acc);
        int col = wave * 16 + (lane & 15);
        float bb = b1[col];
#pragma unroll
        for (int i = 0; i < 8; ++i) {
            int r = i + ((lane >> 4) << 3);
            store_A1(pmid, r, col, fmaxf(acc[i] + bb, 0.0f));  // write bf16 A layout
        }
    }
    __syncthreads();

    v8f acc = {};                        // stage 2: K = 64
#pragma unroll
    for (int kc = 0; kc < 2; ++kc)
        acc = wmma_bf16(load_A(pmid, kc, lane), load_B(w2_v, wave, 2, kc, lane), acc);
    int col = wave * 16 + (lane & 15);
    float bb = b2[col];
#pragma unroll
    for (int i = 0; i < 8; ++i) {
        int r = i + ((lane >> 4) << 3);
        int row = block_row + r;
        if (row < n) cout[(long long)row * 128 + col] = acc[i] + bb;
    }
}

// ---------------------------------------------------------------------------
// Head: z = relu(concat(h,c)@hW1 + hb1);  logits = z@hW2 + hb2   (NC = 1)
// ---------------------------------------------------------------------------
__global__ void head_kernel(const float* __restrict__ h, const float* __restrict__ c,
                            const __bf16* __restrict__ pW1, const float* __restrict__ hb1,
                            const float* __restrict__ hW2, const float* __restrict__ hb2,
                            float* __restrict__ out, int n) {
    __shared__ __bf16 pf[8 * 32 * 16];   // concat tile (16x256), packed A layout
    __shared__ float  lds_z[16 * SROW];  // fp32 z tile for the matvec

    int tid = threadIdx.x;
    int block_row = blockIdx.x * 16;
    for (int i = tid; i < 16 * 32; i += 256) {
        int r = i >> 5, c4 = (i & 31) * 4;
        int row = block_row + r;
        if (row >= n) row = n - 1;
        float4 hv = *(const float4*)&h[(long long)row * 128 + c4];
        float4 cv = *(const float4*)&c[(long long)row * 128 + c4];
        store_A4(pf, r, c4, hv);
        store_A4(pf, r, 128 + c4, cv);
    }
    __syncthreads();

    int lane = tid & 31;
    int wave = tid >> 5;
    const v16bf* w1_v = (const v16bf*)pW1;

    v8f acc = {};
#pragma unroll
    for (int kc = 0; kc < 8; ++kc)       // K = 256
        acc = wmma_bf16(load_A(pf, kc, lane), load_B(w1_v, wave, 8, kc, lane), acc);
    int col = wave * 16 + (lane & 15);
    float bb = hb1[col];
#pragma unroll
    for (int i = 0; i < 8; ++i) {
        int r = i + ((lane >> 4) << 3);
        lds_z[r * SROW + col] = fmaxf(acc[i] + bb, 0.0f);
    }
    __syncthreads();

    // logits matvec (H=128 -> 1), 2 rows per wave
#pragma unroll
    for (int rr = 0; rr < 2; ++rr) {
        int r  = wave * 2 + rr;
        int c0 = lane * 4;
        float s = lds_z[r * SROW + c0 + 0] * hW2[c0 + 0] +
                  lds_z[r * SROW + c0 + 1] * hW2[c0 + 1] +
                  lds_z[r * SROW + c0 + 2] * hW2[c0 + 2] +
                  lds_z[r * SROW + c0 + 3] * hW2[c0 + 3];
#pragma unroll
        for (int m = 16; m >= 1; m >>= 1) s += __shfl_xor(s, m, 32);
        int row = block_row + r;
        if (lane == 0 && row < n) out[row] = s + hb2[0];
    }
}

// ---------------------------------------------------------------------------
// Host-side orchestration
// ---------------------------------------------------------------------------
extern "C" void kernel_launch(void* const* d_in, const int* in_sizes, int n_in,
                              void* d_out, int out_size, void* d_ws, size_t ws_size,
                              hipStream_t stream) {
    const float*     x     = (const float*)d_in[0];
    const long long* ei    = (const long long*)d_in[1];
    const float*     ctx   = (const float*)d_in[2];
    const float*     Wl    = (const float*)d_in[3];
    const float*     bl    = (const float*)d_in[4];
    const float*     Wr    = (const float*)d_in[5];
    const float*     gma   = (const float*)d_in[6];
    const float*     bta   = (const float*)d_in[7];
    const float*     ctxW1 = (const float*)d_in[8];
    const float*     ctxb1 = (const float*)d_in[9];
    const float*     ctxW2 = (const float*)d_in[10];
    const float*     ctxb2 = (const float*)d_in[11];
    const float*     hW1   = (const float*)d_in[12];
    const float*     hb1   = (const float*)d_in[13];
    const float*     hW2   = (const float*)d_in[14];
    const float*     hb2   = (const float*)d_in[15];
    float*           out   = (float*)d_out;

    const int N = in_sizes[0] / 128;
    const int E = in_sizes[1] / 2;

    // Workspace carve-out (256B aligned)
    char*  ws  = (char*)d_ws;
    size_t off = 0;
    auto carve = [&](size_t bytes) {
        void* p = ws + off;
        off = (off + bytes + 255) & ~(size_t)255;
        return p;
    };
    __bf16*       pWl   = (__bf16*)carve((size_t)3 * 128 * 128 * 2);
    __bf16*       pWr   = (__bf16*)carve((size_t)3 * 128 * 128 * 2);
    __bf16*       pcW1  = (__bf16*)carve((size_t)128 * 64 * 2);
    __bf16*       pcW2  = (__bf16*)carve((size_t)64 * 128 * 2);
    __bf16*       phW1  = (__bf16*)carve((size_t)256 * 128 * 2);
    unsigned int* deg   = (unsigned int*)carve((size_t)N * 4);
    float*        invd  = (float*)carve((size_t)N * 4);
    float*        msg   = (float*)carve((size_t)N * 128 * 4);
    float*        hbuf0 = (float*)carve((size_t)N * 128 * 4);
    float*        hbuf1 = (float*)carve((size_t)N * 128 * 4);
    float*        cbuf  = (float*)carve((size_t)N * 128 * 4);

    const int TB = 256;
    const int rowBlocks = (N + 15) / 16;

    // Degrees
    fill_u32_kernel<<<(N + TB - 1) / TB, TB, 0, stream>>>(deg, N);
    degree_kernel<<<(E + TB - 1) / TB, TB, 0, stream>>>(ei, deg, E);
    invdeg_kernel<<<(N + TB - 1) / TB, TB, 0, stream>>>(deg, invd, N);

    // Pack weights to bf16 WMMA-B layout (tiny, once per call)
    auto pack = [&](const float* W, __bf16* P, int K, int Nn) {
        int threads = (Nn / 16) * (K / 32) * 32;
        pack_weight_kernel<<<(threads + TB - 1) / TB, TB, 0, stream>>>(W, P, K, Nn);
    };
    for (int l = 0; l < 3; ++l) {
        pack(Wl + (size_t)l * 128 * 128, pWl + (size_t)l * 128 * 128, 128, 128);
        pack(Wr + (size_t)l * 128 * 128, pWr + (size_t)l * 128 * 128, 128, 128);
    }
    pack(ctxW1, pcW1, 128, 64);
    pack(ctxW2, pcW2, 64, 128);
    pack(hW1,  phW1, 256, 128);

    // 3 SAGE layers
    const float* hin = x;
    float* ha = hbuf0;
    float* hb = hbuf1;
    long long msg4 = (long long)N * 32;
    long long sth  = (long long)E * 32;
    for (int l = 0; l < 3; ++l) {
        fill_f4_kernel<<<(unsigned)((msg4 + TB - 1) / TB), TB, 0, stream>>>((float4*)msg, msg4);
        scatter_kernel<<<(unsigned)((sth + TB - 1) / TB), TB, 0, stream>>>(hin, ei, msg, E);
        sage_layer_kernel<<<rowBlocks, TB, 0, stream>>>(
            hin, msg, invd,
            pWl + (size_t)l * 128 * 128, pWr + (size_t)l * 128 * 128,
            bl + (size_t)l * 128, gma + (size_t)l * 128, bta + (size_t)l * 128,
            ha, N);
        hin = ha;
        float* t = ha; ha = hb; hb = t;
    }

    // ctx MLP + head
    ctx_mlp_kernel<<<rowBlocks, TB, 0, stream>>>(ctx, pcW1, pcW2, ctxb1, ctxb2, cbuf, N);
    head_kernel<<<rowBlocks, TB, 0, stream>>>(hin, cbuf, phW1, hb1, hW2, hb2, out, N);
}
